// ChebNetGCN_31404800868554
// MI455X (gfx1250) — matrix operand order repaired and checked by reading, not compiled, not run on hardware
//
#include <hip/hip_runtime.h>

// ---------------------------------------------------------------------------
// CDNA5 (gfx1250) ChebNet GCN.
//  - SpMM: fp32 global atomics (L2-resident gather/scatter).
//  - Cheb matrix packed once per layer into bf16 WMMA A-fragment tiles.
//  - GEMM: pure global->VGPR fragment loads, no LDS/barriers,
//    4x v_wmma_f32_16x16x32_bf16 per wave per k-step.
// ---------------------------------------------------------------------------

typedef __attribute__((ext_vector_type(16))) __bf16 v16bf;
typedef __attribute__((ext_vector_type(8)))  __bf16 v8bf;
typedef __attribute__((ext_vector_type(8)))  float  v8f;

// ------------------------------- utility kernels ---------------------------

__global__ void cheb_fill_zero(float4* __restrict__ p, long n4) {
    long i = (long)blockIdx.x * blockDim.x + threadIdx.x;
    if (i < n4) p[i] = make_float4(0.f, 0.f, 0.f, 0.f);
}

__global__ void cheb_neg_copy(float4* __restrict__ dst, const float4* __restrict__ src, long n4) {
    long i = (long)blockIdx.x * blockDim.x + threadIdx.x;
    if (i < n4) {
        float4 v = src[i];
        dst[i] = make_float4(-v.x, -v.y, -v.z, -v.w);
    }
}

// ------------------------------- SpMM (COO, atomic) ------------------------
// y[r, :] += scale * w_e * x[c, :]   one wave per edge, lanes over features.

__global__ void cheb_spmm_atomic(const int* __restrict__ rows,
                                 const int* __restrict__ cols,
                                 const float* __restrict__ ew,
                                 const float* __restrict__ x,
                                 float* __restrict__ y,
                                 int E, int F, float scale) {
    int edge = blockIdx.x * (blockDim.x >> 5) + (threadIdx.x >> 5);
    if (edge >= E) return;
    int lane = threadIdx.x & 31;
    int r = rows[edge];
    int c = cols[edge];
    float we = ew[edge] * scale;
    const float4* xs = (const float4*)(x + (size_t)c * F);
    float* yr = y + (size_t)r * F;
    int f4 = F >> 2;
    for (int i = lane; i < f4; i += 32) {
        float4 v = xs[i];
        float* p = yr + (i << 2);
        __hip_atomic_fetch_add(p + 0, we * v.x, __ATOMIC_RELAXED, __HIP_MEMORY_SCOPE_AGENT);
        __hip_atomic_fetch_add(p + 1, we * v.y, __ATOMIC_RELAXED, __HIP_MEMORY_SCOPE_AGENT);
        __hip_atomic_fetch_add(p + 2, we * v.z, __ATOMIC_RELAXED, __HIP_MEMORY_SCOPE_AGENT);
        __hip_atomic_fetch_add(p + 3, we * v.w, __ATOMIC_RELAXED, __HIP_MEMORY_SCOPE_AGENT);
    }
}

// ------------------------- weight repack: fp32 -> bf16 B fragments ---------
// B per-lane order: lane = n | (k'&16), slot = k'&15 (32 lanes x 16 halves).
// Tiles stored k-major: Wp[(nt*ktiles + kt)*512 + lane*16 + slot]

__global__ void cheb_repack_w(const float* __restrict__ W, __bf16* __restrict__ Wp,
                              int FK, int total) {
    int idx = blockIdx.x * blockDim.x + threadIdx.x;
    if (idx >= total) return;
    int slot = idx & 15;
    int lane = (idx >> 4) & 31;
    int tile = idx >> 9;
    int ktiles = FK >> 5;
    int kt = tile % ktiles;
    int nt = tile / ktiles;
    int n  = lane & 15;
    int kk = slot | (lane & 16);
    int h = nt * 16 + n;
    int k = kt * 32 + kk;
    Wp[idx] = (__bf16)W[(size_t)h * FK + k];
}

// ------------------- Cheb matrix pack: fp32 T0/T1/T2 -> bf16 A fragments ---
// cheb[n, 3f+q] from (t0,t1,t2)[n,f].  A per-lane order (contiguous k, same as B):
//   lane = (m&15) | (k'&16),  slot = k'&15
// Slots [0..7] and [8..15] of a lane each hold 8 *consecutive* k values,
// so each thread produces one 16-byte (v8bf) store.
// Tiles stored k-major: Apk[(rowTile*ktiles + kt)*512 + lane*16 + slot]

__global__ void cheb_pack_a(const float* __restrict__ t0,
                            const float* __restrict__ t1,
                            const float* __restrict__ t2,
                            int F, int ktiles,
                            __bf16* __restrict__ Apk, long total8) {
    long idx = (long)blockIdx.x * blockDim.x + threadIdx.x;
    if (idx >= total8) return;
    int  sgrp = (int)(idx & 1);          // slot group: 0 -> slots 0..7, 1 -> 8..15
    int  lane = (int)((idx >> 1) & 31);
    long tile = idx >> 6;
    int  kt      = (int)(tile % ktiles);
    long rowTile = tile / ktiles;
    long node = rowTile * 16 + (lane & 15);
    int  kb = kt * 32 + (lane & 16) + (sgrp << 3);
    v8bf v;
    #pragma unroll
    for (int j = 0; j < 8; ++j) {
        int k = kb + j;
        int f = k / 3;
        int q = k - f * 3;
        const float* s = (q == 0) ? t0 : (q == 1) ? t1 : t2;
        v[j] = (__bf16)s[node * F + f];
    }
    *(v8bf*)(Apk + (tile << 9) + lane * 16 + (sgrp << 3)) = v;
}

// ------------------------------- Cheb GEMM (WMMA) --------------------------
// out[n,h] = relu?( sum_k A[n,k] * W[h,k] + bias[h] )
// Block: 256 thr = 8 waves; tile = 32 nodes x 256 h.
// Wave: 16 nodes x 64 h -> 1 A frag + 4 B frags + 4 WMMAs per 32-k step.
// All fragments loaded contiguously from global (pre-packed); no LDS.

__global__ __launch_bounds__(256)
void cheb_gemm_wmma(const __bf16* __restrict__ Apk,
                    const __bf16* __restrict__ Wp,
                    const float* __restrict__ bias,
                    float* __restrict__ out,
                    int ktiles, int Hout, int do_relu) {
    const int tid  = threadIdx.x;
    const int lane = tid & 31;
    const int wave = tid >> 5;
    const int subtile = wave & 1;       // 16-node group within block
    const int hquad   = wave >> 1;      // 0..3 : 64-col group
    const int nodeBase = blockIdx.x * 32;
    const int hBase    = blockIdx.y * 256 + hquad * 64;

    const __bf16* ap =
        Apk + ((size_t)((nodeBase >> 4) + subtile) * ktiles) * 512 + lane * 16;
    const int nt0 = hBase >> 4;
    const __bf16* bp0 = Wp + ((size_t)(nt0 + 0) * ktiles) * 512 + lane * 16;
    const __bf16* bp1 = Wp + ((size_t)(nt0 + 1) * ktiles) * 512 + lane * 16;
    const __bf16* bp2 = Wp + ((size_t)(nt0 + 2) * ktiles) * 512 + lane * 16;
    const __bf16* bp3 = Wp + ((size_t)(nt0 + 3) * ktiles) * 512 + lane * 16;

    v8f acc0 = {}, acc1 = {}, acc2 = {}, acc3 = {};

    #pragma unroll 2
    for (int kt = 0; kt < ktiles; ++kt) {
        v16bf a  = *(const v16bf*)ap;
        v16bf b0 = *(const v16bf*)bp0;
        v16bf b1 = *(const v16bf*)bp1;
        v16bf b2 = *(const v16bf*)bp2;
        v16bf b3 = *(const v16bf*)bp3;
        // prefetch streaming A a few tiles ahead (global_prefetch_b8)
        __builtin_prefetch((const void*)(ap + 512 * 8), 0, 1);
        ap  += 512;
        bp0 += 512;
        bp1 += 512;
        bp2 += 512;
        bp3 += 512;
        acc0 = __builtin_amdgcn_wmma_f32_16x16x32_bf16(
            false, a, false, b0, (short)0, acc0, false, false);
        acc1 = __builtin_amdgcn_wmma_f32_16x16x32_bf16(
            false, a, false, b1, (short)0, acc1, false, false);
        acc2 = __builtin_amdgcn_wmma_f32_16x16x32_bf16(
            false, a, false, b2, (short)0, acc2, false, false);
        acc3 = __builtin_amdgcn_wmma_f32_16x16x32_bf16(
            false, a, false, b3, (short)0, acc3, false, false);
    }

    // epilogue: C/D layout -> VGPR j holds M=j (lanes 0-15) / M=j+8 (16-31)
    const int ncol = lane & 15;
    const int mro  = (lane >> 4) * 8;
    const int h0 = hBase + ncol;
    const float bb0 = bias[h0];
    const float bb1 = bias[h0 + 16];
    const float bb2 = bias[h0 + 32];
    const float bb3 = bias[h0 + 48];
    #pragma unroll
    for (int j = 0; j < 8; ++j) {
        long node = nodeBase + (subtile << 4) + mro + j;
        float* orow = out + node * Hout + h0;
        float v0 = acc0[j] + bb0;
        float v1 = acc1[j] + bb1;
        float v2 = acc2[j] + bb2;
        float v3 = acc3[j] + bb3;
        if (do_relu) {
            v0 = v0 > 0.f ? v0 : 0.f;
            v1 = v1 > 0.f ? v1 : 0.f;
            v2 = v2 > 0.f ? v2 : 0.f;
            v3 = v3 > 0.f ? v3 : 0.f;
        }
        orow[0]  = v0;
        orow[16] = v1;
        orow[32] = v2;
        orow[48] = v3;
    }
}

// ------------------------------- host driver -------------------------------

extern "C" void kernel_launch(void* const* d_in, const int* in_sizes, int n_in,
                              void* d_out, int out_size, void* d_ws, size_t ws_size,
                              hipStream_t stream) {
    const float* x    = (const float*)d_in[0];
    const int*   rows = (const int*)d_in[1];
    const int*   cols = (const int*)d_in[2];
    const float* ew   = (const float*)d_in[3];
    const float* W1   = (const float*)d_in[4];
    const float* b1   = (const float*)d_in[5];
    const float* W2   = (const float*)d_in[6];
    const float* b2   = (const float*)d_in[7];
    const float* W3   = (const float*)d_in[8];
    const float* b3   = (const float*)d_in[9];
    const float* Wout = (const float*)d_in[10];
    const float* bout = (const float*)d_in[11];

    const int F_IN = 256, H = 512, F_OUT = 256;
    const int N = in_sizes[0] / F_IN;      // 100000
    const int E = in_sizes[1];             // 3200000
    float* out = (float*)d_out;

    // workspace layout
    char* ws = (char*)d_ws;
    const size_t szNH = (size_t)N * H * sizeof(float);
    float*  T1 = (float*)(ws);
    float*  T2 = (float*)(ws + szNH);
    float*  HA = (float*)(ws + 2 * szNH);
    float*  HB = (float*)(ws + 3 * szNH);
    __bf16* Apk = (__bf16*)(ws + 4 * szNH);                 // N * 1536 bf16 max
    __bf16* Wp1 = Apk + (size_t)N * (H * 3);
    __bf16* Wp2 = Wp1 + (size_t)H * (F_IN * 3);
    __bf16* Wp3 = Wp2 + (size_t)H * (H * 3);
    __bf16* Wpo = Wp3 + (size_t)H * (H * 3);

    auto repack = [&](const float* W, __bf16* Wp, int Hd, int FK) {
        int total = Hd * FK;
        cheb_repack_w<<<(total + 255) / 256, 256, 0, stream>>>(W, Wp, FK, total);
    };
    repack(W1,   Wp1, H,     F_IN * 3);
    repack(W2,   Wp2, H,     H * 3);
    repack(W3,   Wp3, H,     H * 3);
    repack(Wout, Wpo, F_OUT, H * 3);

    auto layer = [&](const float* in, int F, const __bf16* Wp, const float* bias,
                     float* outp, int Hdim, int relu) {
        long n4 = (long)N * F / 4;
        int blocks = (int)((n4 + 255) / 256);
        cheb_fill_zero<<<blocks, 256, 0, stream>>>((float4*)T1, n4);
        cheb_spmm_atomic<<<(E + 7) / 8, 256, 0, stream>>>(rows, cols, ew, in, T1, E, F, 1.0f);
        cheb_neg_copy<<<blocks, 256, 0, stream>>>((float4*)T2, (const float4*)in, n4);
        cheb_spmm_atomic<<<(E + 7) / 8, 256, 0, stream>>>(rows, cols, ew, T1, T2, E, F, 2.0f);

        const int FK = 3 * F;
        const int ktiles = FK >> 5;
        long total8 = (long)N * FK / 8;
        cheb_pack_a<<<(int)((total8 + 255) / 256), 256, 0, stream>>>(
            in, T1, T2, F, ktiles, Apk, total8);
        cheb_gemm_wmma<<<dim3(N / 32, Hdim / 256), 256, 0, stream>>>(
            Apk, Wp, bias, outp, ktiles, Hdim, relu);
    };

    layer(x,  F_IN, Wp1, b1,   HB,  H,     1);
    layer(HB, H,    Wp2, b2,   HA,  H,     1);
    layer(HA, H,    Wp3, b3,   HB,  H,     1);
    layer(HB, H,    Wpo, bout, out, F_OUT, 0);
}
